// SAModule_16870631538821
// MI455X (gfx1250) — compile-verified, matrix operand order checked by reading, not run on hardware
//
#include <hip/hip_runtime.h>

// Problem constants (match reference)
#define Bc    16
#define Nn    4096
#define FIN   64
#define Sc    1024
#define Kc    64
#define DIN   67      // F_IN + 3
#define KPAD  96      // DIN padded to multiple of 32 for 16x16x32 WMMA
#define OUTC  128
#define NCTR  (Bc*Sc) // 16384
#define RADQ  0.01f   // R^2

typedef _Float16 v16h __attribute__((ext_vector_type(16)));
typedef _Float16 h8   __attribute__((ext_vector_type(8)));
typedef float    v8f  __attribute__((ext_vector_type(8)));

// Load one 16x32 f16 operand fragment from row-major LDS.
// Per CDNA5 ISA 16-bit A layout: lane L -> row L%16; lanes 0-15 hold K = k..k+7
// and k+16..k+23 (two contiguous 16B chunks); lanes 16-31 the +8 halves.
// B (32x16) mirrors this with "row" = output column of the transposed weight.
__device__ __forceinline__ v16h ld_frag(const _Float16* buf, int stride, int row, int k0) {
    const _Float16* p = buf + row * stride + k0;
    h8 lo = *(const h8*)(p);
    h8 hi = *(const h8*)(p + 16);
    return __builtin_shufflevector(lo, hi, 0,1,2,3,4,5,6,7,8,9,10,11,12,13,14,15);
}

// ---------------- Phase 1: farthest point sampling (1 block / cloud) --------
__global__ __launch_bounds__(256) void fps_kernel(const float* __restrict__ pos,
                                                  int* __restrict__ idx_ws,
                                                  float* __restrict__ posS,
                                                  float* __restrict__ batS) {
    __shared__ float px[Nn], py[Nn], pz[Nn];
    __shared__ float rv[256];
    __shared__ int   ri[256];
    __shared__ int   lastIdx;
    const int b = blockIdx.x, t = threadIdx.x;
    const float* pb = pos + (size_t)b * Nn * 3;
    for (int i = t; i < Nn; i += 256) {
        px[i] = pb[i*3]; py[i] = pb[i*3+1]; pz[i] = pb[i*3+2];
    }
    if (t == 0) { lastIdx = 0; idx_ws[b*Sc] = 0; }
    float mind[16];
    #pragma unroll
    for (int k = 0; k < 16; ++k) mind[k] = 3.4e38f;
    __syncthreads();

    for (int s = 1; s < Sc; ++s) {
        const int last = lastIdx;
        const float lx = px[last], ly = py[last], lz = pz[last];
        float bv = -1.f; int bi = 0;
        #pragma unroll
        for (int k = 0; k < 16; ++k) {
            const int p = t + (k << 8);
            const float dx = px[p]-lx, dy = py[p]-ly, dz = pz[p]-lz;
            const float d2 = dx*dx + dy*dy + dz*dz;
            mind[k] = fminf(mind[k], d2);
            if (mind[k] > bv) { bv = mind[k]; bi = p; }
        }
        rv[t] = bv; ri[t] = bi;
        __syncthreads();
        for (int stp = 128; stp > 0; stp >>= 1) {
            if (t < stp) {
                const float v2 = rv[t+stp]; const int i2 = ri[t+stp];
                if (v2 > rv[t] || (v2 == rv[t] && i2 < ri[t])) { rv[t] = v2; ri[t] = i2; }
            }
            __syncthreads();
        }
        if (t == 0) { lastIdx = ri[0]; idx_ws[b*Sc + s] = ri[0]; }
        __syncthreads();
    }
    for (int s = t; s < Sc; s += 256) {
        const int i = idx_ws[b*Sc + s];
        const int c = b*Sc + s;
        posS[c*3] = px[i]; posS[c*3+1] = py[i]; posS[c*3+2] = pz[i];
        batS[c] = (float)b;
    }
}

// ---------------- Phase 2: ball query (K nearest within radius) -------------
__global__ __launch_bounds__(64) void ball_kernel(const float* __restrict__ pos,
                                                  const float* __restrict__ posS,
                                                  const int* __restrict__ idx_ws,
                                                  int* __restrict__ nbr_ws) {
    __shared__ float cd[Kc * 64];            // cd[slot*64 + tid], conflict-free
    const int t = threadIdx.x;
    const int c = blockIdx.x * 64 + t;
    const int b = c >> 10;                   // S = 1024
    const float cx = posS[c*3], cy = posS[c*3+1], cz = posS[c*3+2];
    const float* pb = pos + (size_t)b * Nn * 3;
    int* mynbr = nbr_ws + (size_t)c * Kc;

    int cnt = 0, wslot = 0; float wval = -1.f;
    for (int n = 0; n < Nn; ++n) {
        const float dx = pb[n*3]-cx, dy = pb[n*3+1]-cy, dz = pb[n*3+2]-cz;
        const float d2 = dx*dx + dy*dy + dz*dz;
        if (d2 <= RADQ) {
            if (cnt < Kc) {
                cd[cnt*64 + t] = d2; mynbr[cnt] = n;
                if (d2 > wval) { wval = d2; wslot = cnt; }
                ++cnt;
            } else if (d2 < wval) {
                cd[wslot*64 + t] = d2; mynbr[wslot] = n;
                wval = -1.f;
                for (int k = 0; k < Kc; ++k) {
                    const float v = cd[k*64 + t];
                    if (v > wval) { wval = v; wslot = k; }
                }
            }
        }
    }
    // Pad with the center's own point: always in-ball (d2=0), idempotent for max-agg
    const int self = idx_ws[c];
    for (int k = cnt; k < Kc; ++k) mynbr[k] = self;
}

// ---------------- Phase 3: WMMA PointNetConv MLP + max aggregation ----------
__global__ __launch_bounds__(128) void conv_kernel(const float* __restrict__ x,
    const float* __restrict__ pos, const float* __restrict__ posS,
    const int* __restrict__ nbr_ws,
    const float* __restrict__ W1, const float* __restrict__ b1,
    const float* __restrict__ W2, const float* __restrict__ b2,
    const float* __restrict__ W3, const float* __restrict__ b3,
    float* __restrict__ outF) {
    __shared__ __align__(16) _Float16 sW1[64 * KPAD];   // W1^T: [n=64][k=96]
    __shared__ __align__(16) _Float16 sW2[64 * 64];     // W2^T: [n=64][k=64]
    __shared__ __align__(16) _Float16 sW3[128 * 64];    // W3^T: [n=128][k=64]
    __shared__ __align__(16) _Float16 sMsg[64 * KPAD];  // msg  : [K=64][96]
    __shared__ __align__(16) _Float16 sH[4 * 16 * 64];  // per-wave h buffer
    __shared__ float sB1[64], sB2[64], sB3[128];
    __shared__ unsigned int uimax[128];

    const int t = threadIdx.x;
    // Stage transposed f16 weights + biases once per block
    for (int i = t; i < 64*KPAD; i += 128) {
        const int n = i / KPAD, k = i - n*KPAD;
        sW1[i] = (k < DIN) ? (_Float16)W1[k*64 + n] : (_Float16)0.f;
    }
    for (int i = t; i < 64*64;  i += 128) { const int n = i>>6, k = i&63; sW2[i] = (_Float16)W2[k*64  + n]; }
    for (int i = t; i < 128*64; i += 128) { const int n = i>>6, k = i&63; sW3[i] = (_Float16)W3[k*128 + n]; }
    if (t < 64)  { sB1[t] = b1[t]; sB2[t] = b2[t]; }
    sB3[t] = b3[t];

    const int lane = t & 31, wv = t >> 5;
    const int col  = lane & 15;                 // column within 16-wide tile
    const int kg   = (lane < 16) ? 0 : 8;       // K lane-group offset
    const int rb   = (lane < 16) ? 0 : 8;       // row base in C/D tile
    _Float16* myh  = sH + wv * 16 * 64;         // this wave's private 16x64 buffer

    for (int c = blockIdx.x; c < NCTR; c += gridDim.x) {
        uimax[t] = 0u;
        // Gather msg = [x_j | pos_j - pos_i | 0-pad] as f16. 2 threads / row.
        {
            const int row = t >> 1, half = t & 1;
            const int b_ = c >> 10;
            const int j  = nbr_ws[(size_t)c * Kc + row];
            const size_t g = (size_t)b_ * Nn + j;
            const float* xp = x + g * FIN + half * 32;
            _Float16* mrow = sMsg + row * KPAD + half * 32;
            #pragma unroll
            for (int q = 0; q < 32; ++q) mrow[q] = (_Float16)xp[q];
            if (half) {
                const float cx = posS[c*3], cy = posS[c*3+1], cz = posS[c*3+2];
                _Float16* m2 = sMsg + row * KPAD;
                m2[64] = (_Float16)(pos[g*3]   - cx);
                m2[65] = (_Float16)(pos[g*3+1] - cy);
                m2[66] = (_Float16)(pos[g*3+2] - cz);
                #pragma unroll
                for (int q = DIN; q < KPAD; ++q) m2[q] = (_Float16)0.f;
            }
        }
        __syncthreads();

        // Layer 1: h1 = relu(msg @ W1 + b1)   [16x96]·[96x64] per wave
        #pragma unroll
        for (int nt = 0; nt < 4; ++nt) {
            const float bb = sB1[nt*16 + col];
            v8f acc = {bb,bb,bb,bb,bb,bb,bb,bb};
            #pragma unroll
            for (int kt = 0; kt < 3; ++kt) {
                v16h A  = ld_frag(sMsg, KPAD, wv*16 + col, kt*32 + kg);
                v16h Bm = ld_frag(sW1,  KPAD, nt*16 + col, kt*32 + kg);
                acc = __builtin_amdgcn_wmma_f32_16x16x32_f16(false, A, false, Bm,
                                                             (short)0, acc, false, false);
            }
            #pragma unroll
            for (int r = 0; r < 8; ++r)
                myh[(r+rb)*64 + nt*16 + col] = (_Float16)fmaxf(acc[r], 0.f);
        }
        // Layer 2: h2 = relu(h1 @ W2 + b2) — accumulate all tiles, then overwrite myh
        {
            v8f acc2[4];
            #pragma unroll
            for (int nt = 0; nt < 4; ++nt) {
                const float bb = sB2[nt*16 + col];
                v8f z = {bb,bb,bb,bb,bb,bb,bb,bb};
                acc2[nt] = z;
            }
            #pragma unroll
            for (int kt = 0; kt < 2; ++kt) {
                v16h A = ld_frag(myh, 64, col, kt*32 + kg);
                #pragma unroll
                for (int nt = 0; nt < 4; ++nt) {
                    v16h Bm = ld_frag(sW2, 64, nt*16 + col, kt*32 + kg);
                    acc2[nt] = __builtin_amdgcn_wmma_f32_16x16x32_f16(false, A, false, Bm,
                                                                      (short)0, acc2[nt], false, false);
                }
            }
            #pragma unroll
            for (int nt = 0; nt < 4; ++nt)
                #pragma unroll
                for (int r = 0; r < 8; ++r)
                    myh[(r+rb)*64 + nt*16 + col] = (_Float16)fmaxf(acc2[nt][r], 0.f);
        }
        // Layer 3: out = max_rows relu(h2 @ W3 + b3); relu(x) max-reduced == max(0, max x)
        #pragma unroll
        for (int nt = 0; nt < 8; ++nt) {
            const float bb = sB3[nt*16 + col];
            v8f acc = {bb,bb,bb,bb,bb,bb,bb,bb};
            #pragma unroll
            for (int kt = 0; kt < 2; ++kt) {
                v16h A  = ld_frag(myh, 64, col, kt*32 + kg);
                v16h Bm = ld_frag(sW3, 64, nt*16 + col, kt*32 + kg);
                acc = __builtin_amdgcn_wmma_f32_16x16x32_f16(false, A, false, Bm,
                                                             (short)0, acc, false, false);
            }
            float m = 0.f;
            #pragma unroll
            for (int r = 0; r < 8; ++r) m = fmaxf(m, acc[r]);
            // values >= 0 -> IEEE order == unsigned order; ds_max_u32 is exact
            atomicMax(&uimax[nt*16 + col], __float_as_uint(m));
        }
        __syncthreads();
        outF[(size_t)c * OUTC + t] = __uint_as_float(uimax[t]);
        __syncthreads();
    }
}

extern "C" void kernel_launch(void* const* d_in, const int* in_sizes, int n_in,
                              void* d_out, int out_size, void* d_ws, size_t ws_size,
                              hipStream_t stream) {
    (void)in_sizes; (void)n_in; (void)out_size; (void)ws_size;
    const float* x   = (const float*)d_in[0];
    const float* pos = (const float*)d_in[1];
    /* d_in[2] = batch (implied by layout: cloud = index / N) */
    const float* W1 = (const float*)d_in[3];
    const float* b1 = (const float*)d_in[4];
    const float* W2 = (const float*)d_in[5];
    const float* b2 = (const float*)d_in[6];
    const float* W3 = (const float*)d_in[7];
    const float* b3 = (const float*)d_in[8];

    float* outF = (float*)d_out;                    // [16384 x 128]
    float* posS = outF + (size_t)NCTR * OUTC;       // [16384 x 3]
    float* batS = posS + (size_t)NCTR * 3;          // [16384]

    int* idx_ws = (int*)d_ws;                       // [16384]
    int* nbr_ws = idx_ws + NCTR;                    // [16384 x 64]

    fps_kernel <<<Bc,       256, 0, stream>>>(pos, idx_ws, posS, batS);
    ball_kernel<<<NCTR/64,   64, 0, stream>>>(pos, posS, idx_ws, nbr_ws);
    conv_kernel<<<512,      128, 0, stream>>>(x, pos, posS, nbr_ws,
                                              W1, b1, W2, b2, W3, b3, outF);
}
